// OMAD_Net_3204045603278
// MI455X (gfx1250) — matrix operand order repaired and checked by reading, not compiled, not run on hardware
//
#include <hip/hip_runtime.h>
#include <string.h>

typedef __attribute__((ext_vector_type(16))) _Float16 v16h;
typedef __attribute__((ext_vector_type(8)))  float    v8f;

#define NPTS 8192
#define BSZ  32

// ---------------- output offsets (floats) ----------------
#define OFF_PPK  524288
#define OFF_QUAT 527360
#define OFF_R    527488
#define OFF_T    527776
#define OFF_JS   527872
#define OFF_BETA 527904
#define OFF_NPK  528224
#define OFF_JV   531296
#define OFF_JU   531392

// Scheduling fence: keep each layer's weight-fragment loads inside its region.
#if __has_builtin(__builtin_amdgcn_sched_barrier)
#define SFENCE() __builtin_amdgcn_sched_barrier(0)
#else
#define SFENCE() asm volatile("" ::: "memory")
#endif

// ---------------- device helpers ----------------
__device__ __forceinline__ v8f wmma32(v16h a, v16h b, v8f c) {
  return __builtin_amdgcn_wmma_f32_16x16x32_f16(false, a, false, b, (short)0, c, false, false);
}
__device__ __forceinline__ v16h ldfrag(const _Float16* f, int idx, int lane) {
  return *reinterpret_cast<const v16h*>(f + (size_t)idx * 512 + lane * 16);
}
// lane p <-> lane p+16 exchange (ds_swizzle SWAPX16: and=0x1f, xor=0x10)
__device__ __forceinline__ float swap16(float v) {
  return __int_as_float(__builtin_amdgcn_ds_swizzle(__float_as_int(v), 0x401F));
}
// one 16-out tile: acc over kc K-chunks, add folded bias, optional relu
__device__ __forceinline__ void tile_acc(const _Float16* frags, int fbase, int t, int kc,
                                         const v16h* inB, const float* bias, bool relu,
                                         int lane, float o[8]) {
  v8f c = {0.f, 0.f, 0.f, 0.f, 0.f, 0.f, 0.f, 0.f};
  for (int k = 0; k < kc; ++k)
    c = wmma32(ldfrag(frags, fbase + t * kc + k, lane), inB[k], c);
  const float* bp = bias + t * 16 + ((lane >= 16) ? 8 : 0);
#pragma unroll
  for (int r = 0; r < 8; ++r) {
    float v = c[r] + bp[r];
    o[r] = relu ? fmaxf(v, 0.f) : v;
  }
}
// convert accumulator pair (out-tiles 2c, 2c+1) into next layer's B fragment.
// B layout: lane L holds col N=L%16, K = e + 16*(L>=16). D layout gives lane L
// channels r+8*(L>=16) of each tile -> half the values come from partner lane.
__device__ __forceinline__ v16h mk_b(const float a0[8], const float a1[8], int lane) {
  union { v16h v; _Float16 h[16]; } u;
  bool lo = lane < 16;
#pragma unroll
  for (int e = 0; e < 8; ++e) {
    float p0 = swap16(a0[e]);
    float p1 = swap16(a1[e]);
    u.h[e]     = (_Float16)(lo ? a0[e] : p1);
    u.h[e + 8] = (_Float16)(lo ? p0 : a1[e]);
  }
  return u.v;
}
// generic head: 128->64 (bn+relu) then 64->16*tiles2 (linear), store to LDS row
__device__ __forceinline__ void run_head(const _Float16* frags, const float* biases,
                                         const v16h FB[4], int f1, int b1o, int f2, int b2o,
                                         int tiles2, int lane, int hi8, float* dstrow, int chbase) {
  v16h H[2];
  for (int c = 0; c < 2; ++c) {
    float a0[8], a1[8];
    tile_acc(frags, f1, 2 * c,     4, FB, biases + b1o, true, lane, a0);
    tile_acc(frags, f1, 2 * c + 1, 4, FB, biases + b1o, true, lane, a1);
    H[c] = mk_b(a0, a1, lane);
    SFENCE();
  }
  for (int t = 0; t < tiles2; ++t) {
    float o[8];
    tile_acc(frags, f2, t, 2, H, biases + b2o, false, lane, o);
#pragma unroll
    for (int r = 0; r < 8; ++r) dstrow[chbase + t * 16 + hi8 + r] = o[r];
    SFENCE();
  }
}

// ---------------- prep: fold BN, build WMMA A-fragments + bias table ----------------
struct PrepP {
  const float* w[13]; const float* b[13]; const float* g[13];
  const float* be[13]; const float* rm[13]; const float* rv[13];
};
__global__ __launch_bounds__(256) void omad_prep(PrepP pp, _Float16* frags, float* biases) {
  int gid = blockIdx.x * blockDim.x + threadIdx.x;
  const int lcin[13]  = {3, 64, 128, 128, 128, 128, 128, 128, 64, 64, 64, 64, 64};
  const int lcout[13] = {64, 128, 128, 64, 64, 64, 64, 64, 2, 96, 32, 1, 1};
  const int lkc[13]   = {1, 2, 4, 4, 4, 4, 4, 4, 2, 2, 2, 2, 2};
  const int lfrag[14] = {0, 4, 20, 52, 68, 84, 100, 116, 132, 134, 146, 150, 152, 154};
  const int lboff[14] = {0, 64, 192, 320, 384, 448, 512, 576, 640, 656, 752, 784, 800, 816};
  if (gid < 154 * 32) {
    int f = gid >> 5, lane = gid & 31;
    int l = 0;
    while (l < 12 && f >= lfrag[l + 1]) ++l;
    int rel = f - lfrag[l];
    int t = rel / lkc[l], k = rel % lkc[l];
    int out = t * 16 + (lane & 15);
    int hi = (lane >> 4) & 1;
    float scale = 1.f;
    if (pp.g[l] && out < lcout[l]) scale = pp.g[l][out] * rsqrtf(pp.rv[l][out] + 1e-5f);
    for (int e = 0; e < 16; ++e) {
      // A-fragment: K = e + 16*hi  (contiguous, matches mk_b B layout)
      int kk = e + (hi ? 16 : 0) + 32 * k;
      float v = 0.f;
      if (out < lcout[l] && kk < lcin[l]) v = pp.w[l][out * lcin[l] + kk] * scale;
      frags[(size_t)f * 512 + lane * 16 + e] = (_Float16)v;
    }
  } else if (gid < 154 * 32 + 1024) {
    int j = gid - 154 * 32;
    float v = 0.f;
    if (j < 816) {
      int l = 0;
      while (l < 12 && j >= lboff[l + 1]) ++l;
      int out = j - lboff[l];
      if (out < lcout[l]) {
        float bb = pp.b[l][out];
        if (pp.g[l]) {
          float inv = pp.g[l][out] * rsqrtf(pp.rv[l][out] + 1e-5f);
          v = (bb - pp.rm[l][out]) * inv + pp.be[l][out];
        } else v = bb;
      }
    }
    biases[j] = v;
  }
}

// ---------------- main fused per-point kernel (WMMA): ONE 16-pt tile per wave ----------------
__global__ __launch_bounds__(64, 1) void omad_main(const float* __restrict__ pts,
                                                   const _Float16* __restrict__ frags,
                                                   const float* __restrict__ biases,
                                                   float* __restrict__ acc,
                                                   float* __restrict__ outCls) {
  __shared__ float featsh[2][16][128];
  __shared__ float headsh[2][16][176];
  __shared__ float ptsh[2][16][3];
  __shared__ float maskf[2][16];
  __shared__ unsigned gmaxb[128];
  __shared__ float fsumb[128];
  __shared__ float denb[32];
  __shared__ float numb[96];
  __shared__ float miscb[3];  // cnt, jaden, jssum
  const int tid = threadIdx.x;
  const int lane = tid & 31, wv = tid >> 5;
  const int b = blockIdx.y;
  const int p = lane & 15;
  const int hi8 = (lane >= 16) ? 8 : 0;
  for (int i = tid; i < 128; i += 64) { gmaxb[i] = 0u; fsumb[i] = 0.f; }
  for (int i = tid; i < 96; i += 64) numb[i] = 0.f;
  if (tid < 32) denb[tid] = 0.f;
  if (tid < 3) miscb[tid] = 0.f;
  __syncthreads();
  enum { F_PN1 = 0, F_PN2 = 4, F_PN3 = 20, F_CLS1 = 52, F_KP1 = 68, F_ATT1 = 84,
         F_JS1 = 100, F_JA1 = 116, F_CLS2 = 132, F_KP2 = 134, F_ATT2 = 146,
         F_JS2 = 150, F_JA2 = 152 };
  enum { B_PN1 = 0, B_PN2 = 64, B_PN3 = 192, B_CLS1 = 320, B_KP1 = 384, B_ATT1 = 448,
         B_JS1 = 512, B_JA1 = 576, B_CLS2 = 640, B_KP2 = 656, B_ATT2 = 752,
         B_JS2 = 784, B_JA2 = 800 };
  float* acc_den = acc;
  float* acc_num = acc + 1024;
  float* acc_jad = acc + 4096;
  float* acc_jss = acc + 4128;
  unsigned* acc_gmx = (unsigned*)(acc + 4160);
  float* acc_fsm = acc + 8256;
  float* acc_cnt = acc + 12352;
  const int tile = blockIdx.x * 2 + wv;  // one tile per wave: no loop -> no LICM spills
  // X as B operand: lane<16 holds point coords at K=0..2; lanes>=16 (K>=16) zero
  union { v16h v; _Float16 h[16]; } xu;
#pragma unroll
  for (int e = 0; e < 16; ++e) xu.h[e] = (_Float16)0.f;
  if (lane < 16) {
    const float* q = pts + ((size_t)b * NPTS + tile * 16 + lane) * 3;
    float x = q[0], y = q[1], z = q[2];
    xu.h[0] = (_Float16)x; xu.h[1] = (_Float16)y; xu.h[2] = (_Float16)z;
    ptsh[wv][lane][0] = x; ptsh[wv][lane][1] = y; ptsh[wv][lane][2] = z;
  }
  v16h B0 = xu.v;
  SFENCE();
  v16h B1[2], B2[4], FB[4];
  for (int c = 0; c < 2; ++c) {               // pn1: 3->64
    float a0[8], a1[8];
    tile_acc(frags, F_PN1, 2 * c,     1, &B0, biases + B_PN1, true, lane, a0);
    tile_acc(frags, F_PN1, 2 * c + 1, 1, &B0, biases + B_PN1, true, lane, a1);
    B1[c] = mk_b(a0, a1, lane);
    SFENCE();
  }
  for (int c = 0; c < 4; ++c) {               // pn2: 64->128
    float a0[8], a1[8];
    tile_acc(frags, F_PN2, 2 * c,     2, B1, biases + B_PN2, true, lane, a0);
    tile_acc(frags, F_PN2, 2 * c + 1, 2, B1, biases + B_PN2, true, lane, a1);
    B2[c] = mk_b(a0, a1, lane);
    SFENCE();
  }
  for (int c = 0; c < 4; ++c) {               // pn3: 128->128 (feat)
    float a0[8], a1[8];
    tile_acc(frags, F_PN3, 2 * c,     4, B2, biases + B_PN3, true, lane, a0);
    tile_acc(frags, F_PN3, 2 * c + 1, 4, B2, biases + B_PN3, true, lane, a1);
#pragma unroll
    for (int r = 0; r < 8; ++r) {
      featsh[wv][p][(2 * c) * 16 + hi8 + r]     = a0[r];
      featsh[wv][p][(2 * c + 1) * 16 + hi8 + r] = a1[r];
    }
    FB[c] = mk_b(a0, a1, lane);
    SFENCE();
  }
  float* drow = &headsh[wv][p][0];
  run_head(frags, biases, FB, F_CLS1, B_CLS1, F_CLS2, B_CLS2, 1, lane, hi8, drow, 0);
  SFENCE();
  run_head(frags, biases, FB, F_KP1,  B_KP1,  F_KP2,  B_KP2,  6, lane, hi8, drow, 16);
  SFENCE();
  run_head(frags, biases, FB, F_ATT1, B_ATT1, F_ATT2, B_ATT2, 2, lane, hi8, drow, 112);
  SFENCE();
  run_head(frags, biases, FB, F_JS1,  B_JS1,  F_JS2,  B_JS2,  1, lane, hi8, drow, 144);
  SFENCE();
  run_head(frags, biases, FB, F_JA1,  B_JA1,  F_JA2,  B_JA2,  1, lane, hi8, drow, 160);
  SFENCE();
  __syncthreads();
  if (lane < 16) {
    float c0 = headsh[wv][lane][0], c1 = headsh[wv][lane][1];
    float m = (c0 >= c1) ? 1.f : 0.f;           // argmax==0
    maskf[wv][lane] = m;
    size_t o = ((size_t)b * NPTS + tile * 16 + lane) * 2;
    outCls[o] = c0; outCls[o + 1] = c1;
    if (m > 0.f) atomicAdd(&miscb[0], 1.f);
    float e = __expf(headsh[wv][lane][160]);     // ja -> softmax-over-n pieces
    atomicAdd(&miscb[1], e);
    atomicAdd(&miscb[2], headsh[wv][lane][144] * e);
  }
  __syncthreads();
  for (int i = lane; i < 16 * 128; i += 32) {    // gfeat max + masked fsum
    int pp2 = i >> 7, ch = i & 127;
    float v = featsh[wv][pp2][ch];               // relu'd, >=0: uint-max works
    atomicMax(&gmaxb[ch], __float_as_uint(v));
    if (maskf[wv][pp2] > 0.f) atomicAdd(&fsumb[ch], v);
  }
  for (int i = lane; i < 512; i += 32) {         // kp softmax numer/denom (SOFT=1)
    int pp2 = i >> 5, k = i & 31;
    float pr = __expf(headsh[wv][pp2][112 + k]);
    atomicAdd(&denb[k], pr);
#pragma unroll
    for (int d = 0; d < 3; ++d) {
      float coord = ptsh[wv][pp2][d] + headsh[wv][pp2][16 + k * 3 + d];
      atomicAdd(&numb[k * 3 + d], pr * coord);
    }
  }
  __syncthreads();
  for (int i = tid; i < 128; i += 64) {
    atomicMax(&acc_gmx[b * 128 + i], gmaxb[i]);
    atomicAdd(&acc_fsm[b * 128 + i], fsumb[i]);
  }
  for (int i = tid; i < 96; i += 64) atomicAdd(&acc_num[b * 96 + i], numb[i]);
  if (tid < 32) atomicAdd(&acc_den[b * 32 + tid], denb[tid]);
  if (tid == 0) {
    atomicAdd(&acc_cnt[b], miscb[0]);
    atomicAdd(&acc_jad[b], miscb[1]);
    atomicAdd(&acc_jss[b], miscb[2]);
  }
}

// ---------------- per-batch tail ----------------
struct TailP {
  const float* r1[6]; const float* r2[6]; const float* t1[6];
  const float* b1[6]; const float* b2[6]; const float* jn1[6]; const float* jn2[6];
  const float *r3w, *r3b, *t2w, *t2b, *b3w, *b3b, *jn3w, *jn3b, *basis, *npl;
};
__device__ __forceinline__ float bnrelu(const float* const* L, int o, float s) {
  float inv = L[2][o] * rsqrtf(L[5][o] + 1e-5f);
  return fmaxf((s + L[1][o] - L[4][o]) * inv + L[3][o], 0.f);
}
__global__ __launch_bounds__(64) void omad_tail(TailP tp, const float* __restrict__ acc,
                                                float* __restrict__ out) {
  __shared__ float bg[128], gf[128], hA[128], hB[64], hC[64], v4s[4], beta10[10], kh[48], kr[48];
  const int b = blockIdx.x, tid = threadIdx.x;
  const float* den = acc;
  const float* num = acc + 1024;
  const float* jad = acc + 4096;
  const float* jss = acc + 4128;
  const unsigned* gmx = (const unsigned*)(acc + 4160);
  const float* fsm = acc + 8256;
  const float* cnt = acc + 12352;
  float cn = cnt[b];
  for (int i = tid; i < 128; i += 64) {
    bg[i] = (cn > 0.f) ? fsm[b * 128 + i] / fmaxf(cn, 1.f) : 0.f;
    gf[i] = __uint_as_float(gmx[b * 128 + i]);
  }
  __syncthreads();
  for (int o = tid; o < 128; o += 64) { float s = 0.f; for (int c = 0; c < 128; ++c) s += tp.r1[0][o * 128 + c] * bg[c]; hA[o] = bnrelu(tp.r1, o, s); }
  __syncthreads();
  if (tid < 64) { float s = 0.f; for (int c = 0; c < 128; ++c) s += tp.r2[0][tid * 128 + c] * hA[c]; hB[tid] = bnrelu(tp.r2, tid, s); }
  __syncthreads();
  if (tid < 4) { float s = 0.f; for (int c = 0; c < 64; ++c) s += tp.r3w[tid * 64 + c] * hB[c]; v4s[tid] = s + tp.r3b[tid]; }
  __syncthreads();
  if (tid == 0) {
    float nq = sqrtf(v4s[0] * v4s[0] + v4s[1] * v4s[1] + v4s[2] * v4s[2] + v4s[3] * v4s[3]);
    float w = v4s[0] / nq, i = v4s[1] / nq, j = v4s[2] / nq, k = v4s[3] / nq;
    float* q = out + OFF_QUAT + b * 4; q[0] = w; q[1] = i; q[2] = j; q[3] = k;
    float* R = out + OFF_R + b * 9;
    R[0] = 1.f - 2.f * (j * j + k * k); R[1] = 2.f * i * j - 2.f * w * k; R[2] = 2.f * w * j + 2.f * i * k;
    R[3] = 2.f * i * j + 2.f * k * w;   R[4] = 1.f - 2.f * (i * i + k * k); R[5] = -2.f * w * i + 2.f * j * k;
    R[6] = -2.f * w * j + 2.f * i * k;  R[7] = 2.f * w * i + 2.f * j * k;   R[8] = 1.f - 2.f * (i * i + j * j);
    out[OFF_JS + b] = jss[b] / jad[b];
  }
  __syncthreads();
  if (tid < 64) { float s = 0.f; for (int c = 0; c < 128; ++c) s += tp.t1[0][tid * 128 + c] * bg[c]; hB[tid] = bnrelu(tp.t1, tid, s); }
  __syncthreads();
  if (tid < 3) { float s = 0.f; for (int c = 0; c < 64; ++c) s += tp.t2w[tid * 64 + c] * hB[c]; out[OFF_T + b * 3 + tid] = s + tp.t2b[tid]; }
  for (int o = tid; o < 128; o += 64) { float s = 0.f; for (int c = 0; c < 128; ++c) s += tp.b1[0][o * 128 + c] * gf[c]; hA[o] = bnrelu(tp.b1, o, s); }
  __syncthreads();
  if (tid < 64) { float s = 0.f; for (int c = 0; c < 128; ++c) s += tp.b2[0][tid * 128 + c] * hA[c]; hC[tid] = bnrelu(tp.b2, tid, s); }
  __syncthreads();
  if (tid < 10) { float s = 0.f; for (int c = 0; c < 64; ++c) s += tp.b3w[tid * 64 + c] * hC[c]; s += tp.b3b[tid];
                  beta10[tid] = s; out[OFF_BETA + b * 10 + tid] = s; }
  __syncthreads();
  if (tid < 48) { int d = tid / 16, m = tid % 16; float s = 0.f;
                  for (int k = 0; k < 10; ++k) s += beta10[k] * tp.basis[(d * 16 + m) * 10 + k]; kh[tid] = s; }
  __syncthreads();
  if (tid < 48) {
    float n0 = tp.npl[0], n1 = tp.npl[1], n2 = tp.npl[2];
    float il = rsqrtf(n0 * n0 + n1 * n1 + n2 * n2);
    float nn[3] = {n0 * il, n1 * il, n2 * il};
    int d = tid / 16, m = tid % 16; float s = 0.f;
    for (int j2 = 0; j2 < 3; ++j2) { float r = ((d == j2) ? 1.f : 0.f) - 2.f * nn[d] * nn[j2]; s += r * kh[j2 * 16 + m]; }
    kr[tid] = s;
  }
  __syncthreads();
  for (int i = tid; i < 96; i += 64) {
    int part = i / 48, rem = i % 48, jj = rem / 3, d = rem % 3;
    out[OFF_NPK + b * 96 + i] = (jj < 8) ? kh[d * 16 + part * 8 + jj] : kr[d * 16 + part * 8 + jj - 8];
  }
  for (int i = tid; i < 96; i += 64) {
    int k = i / 3;
    out[OFF_PPK + b * 96 + i] = num[b * 96 + i] / den[b * 32 + k];
  }
  if (tid < 64) { float s = 0.f; for (int c = 0; c < 10; ++c) s += tp.jn1[0][tid * 10 + c] * beta10[c]; hB[tid] = bnrelu(tp.jn1, tid, s); }
  __syncthreads();
  if (tid < 64) { float s = 0.f; for (int c = 0; c < 64; ++c) s += tp.jn2[0][tid * 64 + c] * hB[c]; hC[tid] = bnrelu(tp.jn2, tid, s); }
  __syncthreads();
  if (tid < 6) { float s = 0.f; for (int c = 0; c < 64; ++c) s += tp.jn3w[tid * 64 + c] * hC[c]; s += tp.jn3b[tid];
    if (tid < 3) out[OFF_JV + b * 3 + tid] = s; else out[OFF_JU + b * 3 + (tid - 3)] = s; }
}

// ---------------- host ----------------
extern "C" void kernel_launch(void* const* d_in, const int* in_sizes, int n_in,
                              void* d_out, int out_size, void* d_ws, size_t ws_size,
                              hipStream_t stream) {
  (void)n_in; (void)out_size; (void)ws_size;
  // layer enum (insertion order): 0..20 params, 21..23 jn1..jn3
  const float* ptr[24][6];
  for (int i = 0; i < 24; ++i) for (int j = 0; j < 6; ++j) ptr[i][j] = nullptr;
  const float* pts = nullptr; const float* basis = nullptr; const float* npl = nullptr;
  static const int bn24[24] = {1,1,1,1,0,1,0,1,0,1,1,0,1,0,1,0,1,0,1,1,0, 1,1,0};
  if (in_sizes[0] == BSZ * NPTS * 3) {
    // dict-insertion order: pts_all, params{w,b,(g,be,rm,rv)}, consts{jn*, basis, n_pl}
    int idx = 0;
    pts = (const float*)d_in[idx++];
    for (int l = 0; l < 21; ++l) {
      ptr[l][0] = (const float*)d_in[idx++];
      ptr[l][1] = (const float*)d_in[idx++];
      if (bn24[l]) { ptr[l][2] = (const float*)d_in[idx++]; ptr[l][3] = (const float*)d_in[idx++];
                     ptr[l][4] = (const float*)d_in[idx++]; ptr[l][5] = (const float*)d_in[idx++]; }
    }
    for (int l = 21; l < 24; ++l) {
      ptr[l][0] = (const float*)d_in[idx++];
      ptr[l][1] = (const float*)d_in[idx++];
      if (bn24[l]) { ptr[l][2] = (const float*)d_in[idx++]; ptr[l][3] = (const float*)d_in[idx++];
                     ptr[l][4] = (const float*)d_in[idx++]; ptr[l][5] = (const float*)d_in[idx++]; }
    }
    basis = (const float*)d_in[idx++];
    npl = (const float*)d_in[idx++];
  } else {
    // sorted pytree order: consts{basis, jn1, jn2, jn3, n_pl}, params (alpha), pts_all
    int idx = 0;
    basis = (const float*)d_in[idx++];
    auto rd = [&](int l) {
      if (bn24[l]) { ptr[l][1] = (const float*)d_in[idx++]; ptr[l][3] = (const float*)d_in[idx++];
                     ptr[l][2] = (const float*)d_in[idx++]; ptr[l][4] = (const float*)d_in[idx++];
                     ptr[l][5] = (const float*)d_in[idx++]; ptr[l][0] = (const float*)d_in[idx++]; }
      else { ptr[l][1] = (const float*)d_in[idx++]; ptr[l][0] = (const float*)d_in[idx++]; }
    };
    rd(21); rd(22); rd(23);
    npl = (const float*)d_in[idx++];
    static const int smap[21] = {7,8,18,19,20,3,4,16,17,14,15,5,6,0,1,2,9,10,11,12,13};
    for (int s = 0; s < 21; ++s) rd(smap[s]);
    pts = (const float*)d_in[idx++];
  }
  // workspace layout: frags(154KB) | biases(4KB @160KB) | accumulators(@164KB)
  _Float16* frags = (_Float16*)d_ws;
  float* biases = (float*)((char*)d_ws + 160 * 1024);
  float* acc = (float*)((char*)d_ws + 164 * 1024);
  hipMemsetAsync(acc, 0, 12384 * sizeof(float), stream);
  static const int fragmap[13] = {0, 1, 2, 3, 5, 7, 14, 16, 4, 6, 8, 15, 17};
  PrepP pp;
  for (int i = 0; i < 13; ++i) {
    int l = fragmap[i];
    pp.w[i] = ptr[l][0]; pp.b[i] = ptr[l][1]; pp.g[i] = ptr[l][2];
    pp.be[i] = ptr[l][3]; pp.rm[i] = ptr[l][4]; pp.rv[i] = ptr[l][5];
  }
  omad_prep<<<24, 256, 0, stream>>>(pp, frags, biases);
  omad_main<<<dim3(256, BSZ), 64, 0, stream>>>(pts, frags, biases, acc, (float*)d_out);
  TailP tp;
  for (int j = 0; j < 6; ++j) {
    tp.r1[j] = ptr[9][j];  tp.r2[j] = ptr[10][j]; tp.t1[j] = ptr[12][j];
    tp.b1[j] = ptr[18][j]; tp.b2[j] = ptr[19][j]; tp.jn1[j] = ptr[21][j]; tp.jn2[j] = ptr[22][j];
  }
  tp.r3w = ptr[11][0]; tp.r3b = ptr[11][1];
  tp.t2w = ptr[13][0]; tp.t2b = ptr[13][1];
  tp.b3w = ptr[20][0]; tp.b3b = ptr[20][1];
  tp.jn3w = ptr[23][0]; tp.jn3b = ptr[23][1];
  tp.basis = basis; tp.npl = npl;
  omad_tail<<<BSZ, 64, 0, stream>>>(tp, acc, (float*)d_out);
}
